// Encoder_Decoder_fc_81621558493667
// MI455X (gfx1250) — compile-verified, matrix operand order checked by reading, not run on hardware
//
#include <hip/hip_runtime.h>
#include <stdint.h>

// Problem sizes (fixed by the reference)
#define Hh    512
#define Bb    256
#define Tt    512
// Partitioning: 16 persistent workgroups, each owns 32 hidden units.
#define NWG   16
#define NH    32
#define NWAVE 16
#define NTHR  (NWAVE * 32)
// LDS weight row pitch in elements: 520*2=1040 bytes; 1040/4 = 260 = 4 (mod 64 banks)
// -> lane l starts 4 banks after lane l-1: conflict-free b128 reads per 16-lane half.
#define ROWP  520
#define LDS_BYTES (128 * ROWP * 2)

typedef __attribute__((ext_vector_type(8)))  float  f32x8;
typedef __attribute__((ext_vector_type(8)))  __bf16 bf16x8;
typedef __attribute__((ext_vector_type(16))) __bf16 bf16x16;

// ---- d_ws layout (bytes) ----
#define WS_WENC  0u
#define WS_WDEC  (2u * 1024u * 1024u)
#define WS_HBUF0 (4u * 1024u * 1024u)
#define WS_HBUF1 (WS_HBUF0 + (unsigned)(Bb * Hh * 2))
#define WS_CTR   (WS_HBUF0 + 2u * (unsigned)(Bb * Hh * 2))

// CDNA5 has a native V_TANH_F32 transcendental (TRANS32 class); use it when the
// builtin exists. sigmoid(x) = 0.5 + 0.5*tanh(x/2) -> 1 trans + 2 VALU ops.
__device__ __forceinline__ float tanh_f(float x) {
#if __has_builtin(__builtin_amdgcn_tanhf)
  return __builtin_amdgcn_tanhf(x);
#else
  float e = __expf(2.0f * x);
  return (e - 1.0f) / (e + 1.0f);
#endif
}
__device__ __forceinline__ float sig_f(float x) {
#if __has_builtin(__builtin_amdgcn_tanhf)
  return 0.5f + 0.5f * __builtin_amdgcn_tanhf(0.5f * x);
#else
  return 1.0f / (1.0f + __expf(-x));
#endif
}

// CDNA5 async global->LDS copy (ASYNCcnt-tracked), per CDNA5 ISA.
__device__ __forceinline__ void async_g2l_b128(uint32_t lds_byte_off, const void* gsrc) {
  asm volatile("global_load_async_to_lds_b128 %0, %1, off"
               :: "v"(lds_byte_off), "v"(gsrc)
               : "memory");
}
__device__ __forceinline__ void wait_async0() {
#if __has_builtin(__builtin_amdgcn_s_wait_asynccnt)
  __builtin_amdgcn_s_wait_asynccnt(0);
#else
  asm volatile("s_wait_asynccnt 0" ::: "memory");
#endif
}

// ---------------- prep kernel: fp32->bf16 weights, init h0/out/counter ----------------
__global__ void prep_kernel(const float* __restrict__ eWhh, const float* __restrict__ dWhh,
                            const float* __restrict__ linb,
                            unsigned short* __restrict__ wencU, unsigned short* __restrict__ wdecU,
                            unsigned short* __restrict__ h0U,
                            float* __restrict__ out, unsigned* __restrict__ ctr) {
  __bf16* wenc = reinterpret_cast<__bf16*>(wencU);
  __bf16* wdec = reinterpret_cast<__bf16*>(wdecU);
  __bf16* h0   = reinterpret_cast<__bf16*>(h0U);
  size_t i = (size_t)blockIdx.x * blockDim.x + threadIdx.x;
  size_t stride = (size_t)gridDim.x * blockDim.x;
  const size_t NWEL = (size_t)4 * Hh * Hh;  // 1,048,576
  for (size_t k = i; k < NWEL; k += stride) {
    wenc[k] = (__bf16)eWhh[k];
    wdec[k] = (__bf16)dWhh[k];
  }
  for (size_t k = i; k < (size_t)Bb * Hh; k += stride) h0[k] = (__bf16)0.0f;
  float lb = linb[0];
  for (size_t k = i; k < (size_t)Bb * Tt; k += stride) out[k] = lb;
  if (i == 0) *ctr = 0u;
}

// ---------------- persistent LSTM kernel ----------------
// grid = 16 WGs x 512 threads. WG w owns hidden units [32w, 32w+32).
// Wave v owns batch rows [16v,16v+16) x its WG's 32 hidden units; c stays in VGPRs.
__global__ void __launch_bounds__(NTHR, 1) lstm_persistent(
    const float* __restrict__ x,
    const float* __restrict__ eWih, const float* __restrict__ eBih, const float* __restrict__ eBhh,
    const float* __restrict__ dWih, const float* __restrict__ dBih, const float* __restrict__ dBhh,
    const float* __restrict__ linW,
    const unsigned short* __restrict__ wencU, const unsigned short* __restrict__ wdecU,
    unsigned short* __restrict__ h0U, unsigned short* __restrict__ h1U,
    float* __restrict__ out, unsigned* __restrict__ ctr) {
  extern __shared__ __bf16 Wlds[];  // [128 rows][ROWP] bf16 = 130 KB (LDS, 320 KB avail)

  const __bf16* wenc = reinterpret_cast<const __bf16*>(wencU);
  const __bf16* wdec = reinterpret_cast<const __bf16*>(wdecU);
  __bf16* hb0 = reinterpret_cast<__bf16*>(h0U);
  __bf16* hb1 = reinterpret_cast<__bf16*>(h1U);

  const int tid   = threadIdx.x;
  const int wave  = tid >> 5;
  const int lane  = tid & 31;
  const int lcol  = lane & 15;   // N-column / M-row within 16-lane half
  const int half  = lane >> 4;   // which K/M half this lane serves
  const int n0    = blockIdx.x * NH;
  const int mbase = wave * 16;
  const int mrow0 = mbase + half * 8;  // accum VGPR r maps to batch row mrow0 + r

  // --- global step barrier (monotonic counter in d_ws) ---
  unsigned epoch = 0;
  auto gbar = [&]() {
    __syncthreads();
    ++epoch;
    if (tid == 0) {
      __threadfence();
      __hip_atomic_fetch_add(ctr, 1u, __ATOMIC_RELEASE, __HIP_MEMORY_SCOPE_AGENT);
      while (__hip_atomic_load(ctr, __ATOMIC_ACQUIRE, __HIP_MEMORY_SCOPE_AGENT) < epoch * NWG)
        __builtin_amdgcn_s_sleep(1);
    }
    __syncthreads();
  };

  // --- load this WG's 128x512 bf16 weight slice into padded LDS rows (async DMA) ---
  auto loadW = [&](const __bf16* wsrc) {
    for (uint32_t c = (uint32_t)tid; c < 128u * 64u; c += NTHR) {
      uint32_t row  = c >> 6;           // 0..127 : gate*32 + local hidden unit
      uint32_t off  = (c & 63u) * 16u;  // byte offset within the 1024-byte row payload
      uint32_t gate = row >> 5, ln = row & 31u;
      const char* src = (const char*)(wsrc + ((size_t)gate * Hh + n0 + ln) * Hh) + off;
      async_g2l_b128(row * (uint32_t)(ROWP * 2) + off, src);
    }
    wait_async0();
    __syncthreads();
  };

  // --- per-lane persistent state ---
  float wih_c[8], b_c[8];     // [gate*2 + sub], sub selects 16-col subtile
  float c0[8], c1[8];         // cell state (sub=0/1), one value per accum row r
  float pw0 = 0.f, pw1 = 0.f; // lin_W for this lane's two hidden columns

  auto loadConsts = [&](const float* Wih, const float* bih, const float* bhh) {
    for (int g4 = 0; g4 < 4; ++g4)
      for (int s = 0; s < 2; ++s) {
        int n = g4 * Hh + n0 + s * 16 + lcol;
        wih_c[g4 * 2 + s] = Wih[n];
        b_c[g4 * 2 + s]   = bih[n] + bhh[n];
      }
    for (int r = 0; r < 8; ++r) { c0[r] = 0.f; c1[r] = 0.f; }
  };

  loadW(wenc);
  loadConsts(eWih, eBih, eBhh);

  for (int g = 0; g < 2 * Tt; ++g) {
    if (g == Tt) {  // phase switch: decoder weights, c resets, h carries over (hT)
      __syncthreads();
      loadW(wdec);
      loadConsts(dWih, dBih, dBhh);
      pw0 = linW[n0 + lcol];
      pw1 = linW[n0 + 16 + lcol];
    }
    const int  t    = (g < Tt) ? g : g - Tt;
    const bool proj = (g >= Tt);
    const __bf16* hread  = (g & 1) ? hb1 : hb0;
    __bf16*       hwrite = (g & 1) ? hb0 : hb1;

    // ---- gates = h @ Whh_slice^T  via v_wmma_f32_16x16x32_bf16 ----
    f32x8 acc[8];
    f32x8 zero = {0.f, 0.f, 0.f, 0.f, 0.f, 0.f, 0.f, 0.f};
#pragma unroll
    for (int j = 0; j < 8; ++j) acc[j] = zero;

    const __bf16* arow = hread + (size_t)(mbase + lcol) * Hh;  // A: row m=mbase+lcol
    for (int kk = 0; kk < 16; ++kk) {
      const int kbase = kk * 32 + half * 8;  // 16-bit A layout: lane half picks K 0..7 / 8..15
      bf16x8 a_lo = *(const bf16x8*)(arow + kbase);
      bf16x8 a_hi = *(const bf16x8*)(arow + kbase + 16);
      bf16x16 afrag = __builtin_shufflevector(a_lo, a_hi,
          0, 1, 2, 3, 4, 5, 6, 7, 8, 9, 10, 11, 12, 13, 14, 15);

      // Preload the 8 B fragments, then issue the 8 independent WMMAs.
      bf16x16 bf[8];
#pragma unroll
      for (int j = 0; j < 8; ++j) {
        const __bf16* brow = Wlds + (size_t)(j * 16 + lcol) * ROWP;
        bf16x8 b_lo = *(const bf16x8*)(brow + kbase);
        bf16x8 b_hi = *(const bf16x8*)(brow + kbase + 16);
        bf[j] = __builtin_shufflevector(b_lo, b_hi,
            0, 1, 2, 3, 4, 5, 6, 7, 8, 9, 10, 11, 12, 13, 14, 15);
      }
#pragma unroll
      for (int j = 0; j < 8; ++j)
        acc[j] = __builtin_amdgcn_wmma_f32_16x16x32_bf16(
            false, afrag, false, bf[j], (short)0, acc[j], false, false);
    }

    // ---- pointwise LSTM update (no cross-lane traffic: gates align per lane) ----
    float xv[8];
#pragma unroll
    for (int r = 0; r < 8; ++r) xv[r] = x[(size_t)(mrow0 + r) * Tt + t];

    float pv[8];
#pragma unroll
    for (int r = 0; r < 8; ++r) {
      float hsum = 0.f;
#pragma unroll
      for (int s = 0; s < 2; ++s) {
        float gi = acc[0 + s][r] + xv[r] * wih_c[0 + s] + b_c[0 + s];
        float gf = acc[2 + s][r] + xv[r] * wih_c[2 + s] + b_c[2 + s];
        float gg = acc[4 + s][r] + xv[r] * wih_c[4 + s] + b_c[4 + s];
        float go = acc[6 + s][r] + xv[r] * wih_c[6 + s] + b_c[6 + s];
        float cprev = s ? c1[r] : c0[r];
        float cn = sig_f(gf) * cprev + sig_f(gi) * tanh_f(gg);
        if (s) c1[r] = cn; else c0[r] = cn;
        float hv = sig_f(go) * tanh_f(cn);
        hwrite[(size_t)(mrow0 + r) * Hh + (n0 + s * 16 + lcol)] = (__bf16)hv;
        hsum += (s ? pw1 : pw0) * hv;
      }
      pv[r] = hsum;
    }

    // ---- decoder projection: reduce across the 16-lane half, accumulate into out ----
    if (proj) {
#pragma unroll
      for (int r = 0; r < 8; ++r) {
        float v = pv[r];
        v += __shfl_xor(v, 1, 32);
        v += __shfl_xor(v, 2, 32);
        v += __shfl_xor(v, 4, 32);
        v += __shfl_xor(v, 8, 32);
        if (lcol == 0) atomicAdd(&out[(size_t)(mrow0 + r) * Tt + t], v);
      }
    }

    gbar();  // h_t fully published before any WG starts step t+1
  }
}

extern "C" void kernel_launch(void* const* d_in, const int* in_sizes, int n_in,
                              void* d_out, int out_size, void* d_ws, size_t ws_size,
                              hipStream_t stream) {
  const float* x    = (const float*)d_in[0];
  const float* eWih = (const float*)d_in[1];
  const float* eWhh = (const float*)d_in[2];
  const float* eBih = (const float*)d_in[3];
  const float* eBhh = (const float*)d_in[4];
  const float* dWih = (const float*)d_in[5];
  const float* dWhh = (const float*)d_in[6];
  const float* dBih = (const float*)d_in[7];
  const float* dBhh = (const float*)d_in[8];
  const float* linW = (const float*)d_in[9];
  const float* linb = (const float*)d_in[10];
  float* out = (float*)d_out;

  char* ws = (char*)d_ws;
  unsigned short* wenc = (unsigned short*)(ws + WS_WENC);
  unsigned short* wdec = (unsigned short*)(ws + WS_WDEC);
  unsigned short* hb0  = (unsigned short*)(ws + WS_HBUF0);
  unsigned short* hb1  = (unsigned short*)(ws + WS_HBUF1);
  unsigned* ctr        = (unsigned*)(ws + WS_CTR);

  // 130 KB dynamic LDS for the per-WGP padded weight slice (CDNA5 allows up to 320 KB)
  hipFuncSetAttribute((const void*)lstm_persistent,
                      hipFuncAttributeMaxDynamicSharedMemorySize, LDS_BYTES);

  prep_kernel<<<1024, 256, 0, stream>>>(eWhh, dWhh, linb, wenc, wdec, hb0, out, ctr);
  lstm_persistent<<<NWG, NTHR, LDS_BYTES, stream>>>(
      x, eWih, eBih, eBhh, dWih, dBih, dBhh, linW, wenc, wdec, hb0, hb1, out, ctr);
}
